// CRSDBlock_46093589020912
// MI455X (gfx1250) — matrix operand order repaired and checked by reading, not compiled, not run on hardware
//
#include <hip/hip_runtime.h>

// ---------------------------------------------------------------------------
// CRSD reservoir RNN for MI455X (gfx1250), bf16 WMMA, LDS-resident weights.
//
// Shapes: B=32, T=1024, D=R=1024, LAYERS=2, ALPHA=0.1.
// Per step:  r' = 0.9 r + 0.1 tanh(x Wxr + h Whr)
//            h' = tanh(x Wxh + h Whh + r' Wrh)
//
// Strategy:
//  * 64 persistent workgroups; WG g owns columns [16g,16g+16) of r AND of h.
//  * Its 5 weight slices (1024x16 each, bf16) = 160 KB live in LDS for a
//    whole layer -> zero per-step weight traffic (LDS, not L2/HBM bound).
//  * M = 32 batch rows = two 16-row WMMA M-tiles; 4 waves/WG:
//    wave = (m_tile, k_half); K split 2-ways, reduced via LDS.
//  * Two device-scope split barriers per timestep (publish r', publish h').
//  * Weights pre-packed into WMMA B-fragment-major layout; activations kept
//    bf16 row-major so A fragments are two contiguous 16B loads per lane.
// ---------------------------------------------------------------------------

typedef __bf16 v8bf  __attribute__((ext_vector_type(8)));
typedef __bf16 v16bf __attribute__((ext_vector_type(16)));
typedef float  v8f   __attribute__((ext_vector_type(8)));

#define BB   32
#define TT   1024
#define DD   1024
#define NWG  64
#define TPB  128           // 4 waves (wave32)

// LDS layout (bytes)
#define WLDS_BYTES  (5 * 32 * 32 * 16 * 2)          // 163840: [mat][kb][lane][16] bf16
#define PART_OFF    WLDS_BYTES                       // 2048: [m][lane][8] f32 partials
#define RST_OFF     (WLDS_BYTES + 2048)              // 2048: r-state [32 rows][16 cols] f32
#define SMEM_BYTES  (WLDS_BYTES + 2048 + 2048)       // 167936 <= 320 KB/WGP

// Workspace layout (bytes)
#define WS_SYNC_OFF   0
#define WS_WPACK_OFF  256
#define WPACK_BYTES   (2ull * 5 * 1024 * 1024 * 2)   // 20 MB packed bf16 weights
#define WS_XBF_OFF    (WS_WPACK_OFF + WPACK_BYTES)
#define XBF_BYTES     (32ull * 1024 * 1024 * 2)      // 64 MB x_seq in bf16
#define WS_H1_OFF     (WS_XBF_OFF + XBF_BYTES)
#define H1_BYTES      (32ull * 1024 * 1024 * 2)      // 64 MB layer-1 outputs (bf16)
#define WS_HBUF_OFF   (WS_H1_OFF + H1_BYTES)
#define HBUF_BYTES    (2ull * BB * DD * 2)           // double-buffered h (bf16)
#define WS_RPUB_OFF   (WS_HBUF_OFF + HBUF_BYTES)
#define RPUB_BYTES    ((size_t)BB * DD * 2)          // published r' (bf16)

// ---------------------------------------------------------------------------
__global__ void init_sync_kernel(int* s) {
    if (threadIdx.x < 64) s[threadIdx.x] = 0;
}

// f32 -> bf16 bulk convert (x_seq)
__global__ void cvt_bf16_kernel(const float* __restrict__ src,
                                __bf16* __restrict__ dst, long n) {
    long i = (long)blockIdx.x * blockDim.x + threadIdx.x;
    long stride = (long)gridDim.x * blockDim.x;
    for (; i < n; i += stride) dst[i] = (__bf16)src[i];
}

// Pack 5 weight matrices per layer into WMMA-B-fragment-major bf16:
// wpack[layer][mat][g][kb][lane][e], mat order: 0=Wxr 1=Whr 2=Wxh 3=Whh 4=Wrh.
// B fragment (32x16, bf16): N=lane&15; element e -> K = (lane>=16 ? 8 : 0)
// + (e & 7) + (e >= 8 ? 16 : 0) (same k order as the A fragment).
__global__ void pack_weights_kernel(const float* __restrict__ Wxh,
                                    const float* __restrict__ Whh,
                                    const float* __restrict__ Wrh,
                                    const float* __restrict__ Wxr,
                                    const float* __restrict__ Whr,
                                    __bf16* __restrict__ wpack) {
    size_t idx = (size_t)blockIdx.x * blockDim.x + threadIdx.x;
    const size_t total = 2ull * 5 * 1024 * 1024;
    if (idx >= total) return;
    int e    = (int)(idx & 15);
    int lane = (int)((idx >> 4) & 31);
    int kb   = (int)((idx >> 9) & 31);
    int g    = (int)((idx >> 14) & 63);
    int lm   = (int)(idx >> 20);         // layer*5 + mat
    int mat  = lm % 5;
    int l    = lm / 5;
    int k = kb * 32 + ((lane >> 4) << 3) + (e & 7) + ((e >> 3) << 4);
    int n = g * 16 + (lane & 15);
    const float* M = (mat == 0) ? Wxr : (mat == 1) ? Whr
                   : (mat == 2) ? Wxh : (mat == 3) ? Whh : Wrh;
    wpack[idx] = (__bf16)M[((size_t)l * 1024 + k) * 1024 + n];
}

// ---------------------------------------------------------------------------
__device__ __forceinline__ void grid_barrier(int* cnt, volatile int* gen) {
    __syncthreads();
    if (threadIdx.x == 0) {
        __threadfence();                       // release my stores (device scope)
        int g = *gen;
        if (atomicAdd(cnt, 1) == NWG - 1) {
            *cnt = 0;
            __threadfence();
            atomicAdd((int*)gen, 1);
        } else {
            while (*gen == g) { __builtin_amdgcn_s_sleep(2); }
        }
        __threadfence();                       // acquire others' stores
    }
    __syncthreads();
}

// A fragment (16x32 bf16, ISA 7.12.2): lane holds row M = m*16 + (lane&15);
// elements 0..7 -> K = k0..k0+7, elements 8..15 -> K = k0+16..k0+23, where
// k0 = kb*32 + (lane>=16 ? 8 : 0). Both chunks are contiguous 16B loads.
__device__ __forceinline__ v16bf a_frag(const __bf16* __restrict__ rowbase, int k0) {
    v8bf lo = *(const v8bf*)(rowbase + k0);
    v8bf hi = *(const v8bf*)(rowbase + k0 + 16);
    return __builtin_shufflevector(lo, hi, 0, 1, 2, 3, 4, 5, 6, 7,
                                          8, 9, 10, 11, 12, 13, 14, 15);
}

// One 16x512x16 GEMM slice (one K half): 16 k-blocks of v_wmma_f32_16x16x32_bf16,
// B fragments streamed from LDS (contiguous 32B per lane).
__device__ __forceinline__ void mm_slice(v8f& acc,
                                         const __bf16* __restrict__ rowbase,
                                         const __bf16* __restrict__ wlds,
                                         int mat, int kb0, int lane, int koff) {
#pragma unroll 4
    for (int kb = kb0; kb < kb0 + 16; ++kb) {
        v16bf a = a_frag(rowbase, kb * 32 + koff);
        const v16bf* bp =
            (const v16bf*)(wlds + (size_t)mat * 16384 + (size_t)kb * 512 + lane * 16);
        acc = __builtin_amdgcn_wmma_f32_16x16x32_bf16(
            /*neg_a=*/false, a, /*neg_b=*/false, *bp,
            /*c_mod=*/(short)0, acc, /*reuse_a=*/false, /*reuse_b=*/false);
    }
}

// ---------------------------------------------------------------------------
__global__ __launch_bounds__(TPB, 1)
void crsd_main_kernel(const __bf16* __restrict__ xbf,     // [B][T][D] bf16
                      const __bf16* __restrict__ wpack,   // packed weights
                      __bf16* __restrict__ h1seq,         // [B][T][D] bf16 (layer-1 out)
                      __bf16* __restrict__ hbuf,          // [2][B][D] bf16
                      __bf16* __restrict__ rpub,          // [B][D] bf16
                      float*  __restrict__ dout,          // [B][T][D] f32
                      int* sync) {
    extern __shared__ char smem[];
    __bf16* wlds   = (__bf16*)smem;
    float*  part   = (float*)(smem + PART_OFF);
    float*  rstate = (float*)(smem + RST_OFF);

    const int wg   = blockIdx.x;
    const int tid  = threadIdx.x;
    const int lane = tid & 31;          // wave32
    const int wid  = tid >> 5;
    const int m    = wid & 1;           // M tile (batch rows m*16 .. m*16+15)
    const int kh   = wid >> 1;          // K half (0: K<512, 1: K>=512)
    const int kb0  = kh * 16;
    const int n0   = wg * 16;           // owned column base (same for r and h)
    const int colL = lane & 15;
    const int rowA = m * 16 + colL;     // A-fragment batch row for this lane
    const int koff = ((lane >> 4) & 1) * 8;

    int* cnt = sync;
    volatile int* gen = sync + 1;

    for (int layer = 0; layer < 2; ++layer) {
        // ---- load this layer's 5 weight slices into LDS (160 KB, once) ----
        for (int mat = 0; mat < 5; ++mat) {
            const uint4* src = (const uint4*)wpack +
                ((size_t)(layer * 5 + mat) * NWG + wg) * 2048;
            uint4* dst = (uint4*)wlds + (size_t)mat * 2048;
            for (int i = tid; i < 2048; i += TPB) dst[i] = src[i];
        }
        for (int i = tid; i < 512; i += TPB) rstate[i] = 0.0f;   // r = 0
        {   // h double buffers = 0 (grid-cooperative)
            int gt = wg * TPB + tid;
            for (int i = gt; i < 2 * BB * DD; i += NWG * TPB)
                hbuf[i] = (__bf16)0.0f;
        }
        __syncthreads();
        grid_barrier(cnt, gen);

        const __bf16* xin = (layer == 0) ? xbf : h1seq;
        const __bf16* xrowbase = xin + (size_t)rowA * TT * DD;
        int p = 0;

        for (int t = 0; t < TT; ++t) {
            const __bf16* xrow = xrowbase + (size_t)t * DD;
            const __bf16* hrow = hbuf + (size_t)p * BB * DD + (size_t)rowA * DD;
            if (t + 1 < TT) __builtin_prefetch(xrow + DD, 0, 1);

            // ================= Phase A: owned r-columns ===================
            v8f acc = {0.f, 0.f, 0.f, 0.f, 0.f, 0.f, 0.f, 0.f};
            mm_slice(acc, xrow, wlds, /*Wxr*/0, kb0, lane, koff);
            mm_slice(acc, hrow, wlds, /*Whr*/1, kb0, lane, koff);

            if (kh == 1) {
                float* pp = part + (size_t)(m * 32 + lane) * 8;
#pragma unroll
                for (int i = 0; i < 8; ++i) pp[i] = acc[i];
            }
            __syncthreads();
            if (kh == 0) {
                const float* pp = part + (size_t)(m * 32 + lane) * 8;
#pragma unroll
                for (int i = 0; i < 8; ++i) {
                    // C/D layout: lanes<16 -> M=i, lanes>=16 -> M=8+i; N=lane&15
                    int row = m * 16 + ((lane < 16) ? i : 8 + i);
                    float u  = acc[i] + pp[i];
                    float rn = 0.9f * rstate[row * 16 + colL] + 0.1f * tanhf(u);
                    rstate[row * 16 + colL] = rn;
                    rpub[(size_t)row * DD + n0 + colL] = (__bf16)rn;
                }
            }
            __syncthreads();
            grid_barrier(cnt, gen);            // r' fully published

            // ================= Phase B: owned h-columns ===================
            v8f acc2 = {0.f, 0.f, 0.f, 0.f, 0.f, 0.f, 0.f, 0.f};
            mm_slice(acc2, xrow, wlds, /*Wxh*/2, kb0, lane, koff);
            mm_slice(acc2, hrow, wlds, /*Whh*/3, kb0, lane, koff);
            const __bf16* rrow = rpub + (size_t)rowA * DD;
            mm_slice(acc2, rrow, wlds, /*Wrh*/4, kb0, lane, koff);

            if (kh == 1) {
                float* pp = part + (size_t)(m * 32 + lane) * 8;
#pragma unroll
                for (int i = 0; i < 8; ++i) pp[i] = acc2[i];
            }
            __syncthreads();
            if (kh == 0) {
                const float* pp = part + (size_t)(m * 32 + lane) * 8;
#pragma unroll
                for (int i = 0; i < 8; ++i) {
                    int row = m * 16 + ((lane < 16) ? i : 8 + i);
                    float hv = tanhf(acc2[i] + pp[i]);
                    hbuf[(size_t)(p ^ 1) * BB * DD + (size_t)row * DD + n0 + colL] =
                        (__bf16)hv;
                    size_t o = ((size_t)row * TT + t) * DD + n0 + colL;
                    if (layer == 0) h1seq[o] = (__bf16)hv;
                    else            dout[o]  = hv;
                }
            }
            __syncthreads();
            grid_barrier(cnt, gen);            // h' fully published
            p ^= 1;
        }
    }
}

// ---------------------------------------------------------------------------
extern "C" void kernel_launch(void* const* d_in, const int* in_sizes, int n_in,
                              void* d_out, int out_size, void* d_ws, size_t ws_size,
                              hipStream_t stream) {
    const float* x   = (const float*)d_in[0];   // [B,T,D]
    const float* Wxh = (const float*)d_in[1];   // [L,D,D]
    const float* Whh = (const float*)d_in[2];   // [L,D,D]
    const float* Wrh = (const float*)d_in[3];   // [L,R,D]
    const float* Wxr = (const float*)d_in[4];   // [L,D,R]
    const float* Whr = (const float*)d_in[5];   // [L,D,R]
    float* out = (float*)d_out;

    char* ws = (char*)d_ws;
    int*    sync  = (int*)(ws + WS_SYNC_OFF);
    __bf16* wpack = (__bf16*)(ws + WS_WPACK_OFF);
    __bf16* xbf   = (__bf16*)(ws + WS_XBF_OFF);
    __bf16* h1seq = (__bf16*)(ws + WS_H1_OFF);
    __bf16* hbuf  = (__bf16*)(ws + WS_HBUF_OFF);
    __bf16* rpub  = (__bf16*)(ws + WS_RPUB_OFF);

    init_sync_kernel<<<1, 64, 0, stream>>>(sync);

    const long xn = (long)BB * TT * DD;
    cvt_bf16_kernel<<<4096, 256, 0, stream>>>(x, xbf, xn);

    const size_t wtotal = 2ull * 5 * 1024 * 1024;
    pack_weights_kernel<<<(unsigned)((wtotal + 255) / 256), 256, 0, stream>>>(
        Wxh, Whh, Wrh, Wxr, Whr, wpack);

    // Persistent grid: 64 WGs (one per WGP, LDS-limited at 164 KB each).
    crsd_main_kernel<<<NWG, TPB, SMEM_BYTES, stream>>>(
        xbf, wpack, h1seq, hbuf, rpub, out, sync);
}
